// HAN_29712583754248
// MI455X (gfx1250) — matrix operand order repaired and checked by reading, not compiled, not run on hardware
//
#include <hip/hip_runtime.h>
#include <math.h>

// ---------------------------------------------------------------------------
// HAN fact-verification pipeline for MI455X (gfx1250), wave32 + WMMA bf16.
// All GEMMs use v_wmma_f32_16x16x32_bf16 with f32 accumulation, 32x64 wave
// tiles, and register double-buffering of A/B fragments across k-tiles so the
// WMMA pipe overlaps with global_load_b128 traffic (partial loadcnt waits).
// fp32 A operands (the GRU input projection) are converted in-register via
// v_cvt_pk_bf16_f32; the k<300 bounds guard is taken on a *uniform* per-tile
// branch so only the tail k-tile pays divergent EXEC-masked loads.
// ---------------------------------------------------------------------------

typedef __bf16 bf16_t;
typedef __attribute__((ext_vector_type(16))) __bf16        v16bf;
typedef __attribute__((ext_vector_type(8)))  float         v8f;
typedef __attribute__((ext_vector_type(4)))  unsigned int  u32x4;
typedef __attribute__((ext_vector_type(4)))  float         f32x4;

#define E_DIM   300
#define E_PAD   320      // 300 padded to multiple of 32 (weights zero-padded)
#define H_DIM   256
#define G3      768      // 3*H gates (r,z,n)
#define NS      2048     // number of sentences
#define TLEN    64       // sequence length

union FragAB { u32x4 q[2]; v16bf v; };   // 32B WMMA operand fragment

__device__ __forceinline__ float sigmoidf_(float x) { return 1.f / (1.f + expf(-x)); }

// ---------------------------------------------------------------------------
// A-fragment loader (16x32 bf16 tile slice for one lane).
// ISA 7.12.2: lane(row = lane&15, half = lane>>4) holds k = kb+half*8 .. +7
// (elements 0..7) and k = kb+16+half*8 .. +7 (elements 8..15).
// AF32=false: A is bf16 row-major -> two b128 loads.
// AF32=true : A is fp32 (strided sentence slice); convert in-register.
//   `full` is wave-uniform (kb+32 <= kvalid): fast path has NO guards; only
//   the single tail tile takes the divergent per-float4 guarded path
//   (exact: kvalid=300 is a multiple of 4, chunks are 4-aligned).
// ---------------------------------------------------------------------------
template <bool AF32>
__device__ __forceinline__ FragAB load_a_frag(const void* __restrict__ A,
                                              long rowoff, int kb, int half,
                                              int kvalid, bool full) {
    FragAB a;
    int koff = kb + half * 8;
    if constexpr (!AF32) {
        const bf16_t* p = (const bf16_t*)A + rowoff + koff;
        a.q[0] = *reinterpret_cast<const u32x4*>(p);
        a.q[1] = *reinterpret_cast<const u32x4*>(p + 16);
    } else {
        const float* p = (const float*)A + rowoff;
        f32x4 f[4];
        if (full) {                       // uniform branch: s_cbranch, no EXEC mask
#pragma unroll
            for (int c = 0; c < 4; ++c) {
                int ks = koff + (c >= 2 ? 16 : 0) + (c & 1) * 4;
                f[c] = *reinterpret_cast<const f32x4*>(p + ks);
            }
        } else {                          // tail tile only
            f32x4 fz = {0.f, 0.f, 0.f, 0.f};
#pragma unroll
            for (int c = 0; c < 4; ++c) {
                int ks = koff + (c >= 2 ? 16 : 0) + (c & 1) * 4;
                f[c] = (ks < kvalid) ? *reinterpret_cast<const f32x4*>(p + ks) : fz;
            }
        }
#pragma unroll
        for (int e = 0; e < 8; ++e) {
            a.v[e]     = (bf16_t)f[e >> 2][e & 3];
            a.v[8 + e] = (bf16_t)f[2 + (e >> 2)][e & 3];
        }
    }
    return a;
}

// ---------------------------------------------------------------------------
// Generic WMMA GEMM:  C[M][N] = A[M][K] * W[N][K]^T (+ bias[N])
// Block = 128 threads (4 waves); block tile 128 rows x 64 cols.
// Wave tile = 32 rows x 64 cols: 2 A frags + 4 B frags -> 8 WMMAs per k-tile.
// K is always an even multiple of 32 here, so the k-loop ping-pongs two
// register fragment buffers (software pipeline, no remainder).
// B fragment (32x16): lane(col = lane&15, half) reads k = kb+half*8 .. +7 and kb+16+half*8 .. +7
// from the row-major weight row `col` (two b128 loads).
// ---------------------------------------------------------------------------
template <bool AF32>
__global__ __launch_bounds__(128)
void gemm_wmma(const void* __restrict__ A, long lda, int kvalid,
               const bf16_t* __restrict__ W, int ldw,
               const float* __restrict__ bias,
               float* __restrict__ C, int ldc, int K) {
    const int lane  = threadIdx.x & 31;
    const int wave  = threadIdx.x >> 5;
    const int half  = lane >> 4;
    const int r16   = lane & 15;
    const int rbase = blockIdx.y * 128 + wave * 32;
    const int cbase = blockIdx.x * 64;

    const long rowoff0 = (long)(rbase + r16) * lda;
    const long rowoff1 = (long)(rbase + 16 + r16) * lda;

    v8f zero = {0.f, 0.f, 0.f, 0.f, 0.f, 0.f, 0.f, 0.f};
    v8f acc0[4], acc1[4];
#pragma unroll
    for (int c = 0; c < 4; ++c) { acc0[c] = zero; acc1[c] = zero; }

    FragAB abuf[2][2];
    FragAB bbuf[2][4];

    auto load_tile = [&](int buf, int kb) {
        bool full = (kb + 32) <= kvalid;   // wave-uniform
        abuf[buf][0] = load_a_frag<AF32>(A, rowoff0, kb, half, kvalid, full);
        abuf[buf][1] = load_a_frag<AF32>(A, rowoff1, kb, half, kvalid, full);
#pragma unroll
        for (int c = 0; c < 4; ++c) {
            const bf16_t* wrow =
                W + (long)(cbase + c * 16 + r16) * ldw + kb + half * 8;
            bbuf[buf][c].q[0] = *reinterpret_cast<const u32x4*>(wrow);
            bbuf[buf][c].q[1] = *reinterpret_cast<const u32x4*>(wrow + 16);
        }
    };
    auto wmma_tile = [&](int buf) {
#pragma unroll
        for (int c = 0; c < 4; ++c) {
            acc0[c] = __builtin_amdgcn_wmma_f32_16x16x32_bf16(
                false, abuf[buf][0].v, false, bbuf[buf][c].v, (short)0, acc0[c],
                false, false);
            acc1[c] = __builtin_amdgcn_wmma_f32_16x16x32_bf16(
                false, abuf[buf][1].v, false, bbuf[buf][c].v, (short)0, acc1[c],
                false, false);
        }
    };

    load_tile(0, 0);
    for (int kb = 0; kb < K; kb += 64) {
        load_tile(1, kb + 32);          // prefetch next k-tile into buf1
        wmma_tile(0);                   // compute on buf0 while buf1 in flight
        if (kb + 64 < K) load_tile(0, kb + 64);
        wmma_tile(1);
    }

    // C/D layout: VGPR v -> row = rowtile + v + 8*half, col = ctile + (lane&15)
#pragma unroll
    for (int c = 0; c < 4; ++c) {
        int ncol = cbase + c * 16 + r16;
        float bv = bias ? bias[ncol] : 0.f;
#pragma unroll
        for (int v = 0; v < 8; ++v) {
            int row0 = rbase + v + half * 8;
            C[(long)row0 * ldc + ncol]        = acc0[c][v] + bv;
            C[(long)(row0 + 16) * ldc + ncol] = acc1[c][v] + bv;
        }
    }
}

// ---------------------------------------------------------------------------
// Generic f32->bf16 cast with row re-stride / column pad / dst offset.
// ---------------------------------------------------------------------------
__global__ void cast_f32_bf16(const float* __restrict__ src, int srcStride,
                              bf16_t* __restrict__ dst, int dstStride, int dstOff,
                              int rows, int copyCols, int totalCols) {
    long idx = (long)blockIdx.x * blockDim.x + threadIdx.x;
    long total = (long)rows * totalCols;
    if (idx >= total) return;
    int r = (int)(idx / totalCols), c = (int)(idx % totalCols);
    float v = (c < copyCols) ? src[(long)r * srcStride + c] : 0.f;
    dst[(long)r * dstStride + dstOff + c] = (bf16_t)v;
}

// ---------------------------------------------------------------------------
// Claim GRU: batch 1, T=64, scalar single-workgroup (27 MFLOP total).
// ---------------------------------------------------------------------------
__global__ __launch_bounds__(256)
void claim_gru(const float* __restrict__ x, const float* __restrict__ Wih,
               const float* __restrict__ Whh, const float* __restrict__ bih,
               const float* __restrict__ bhh, float* __restrict__ hc_out) {
    __shared__ float hbuf[2][H_DIM];
    int j = threadIdx.x;
    hbuf[0][j] = 0.f;
    __syncthreads();
    for (int t = 0; t < TLEN; ++t) {
        int cur = t & 1, nxt = cur ^ 1;
        const float* xt = x + t * E_DIM;
        float ir = bih[j], iz = bih[H_DIM + j], in_ = bih[2 * H_DIM + j];
        for (int k = 0; k < E_DIM; ++k) {
            float xv = xt[k];
            ir  += Wih[(long)j * E_DIM + k] * xv;
            iz  += Wih[(long)(H_DIM + j) * E_DIM + k] * xv;
            in_ += Wih[(long)(2 * H_DIM + j) * E_DIM + k] * xv;
        }
        float hr = bhh[j], hz = bhh[H_DIM + j], hn = bhh[2 * H_DIM + j];
        for (int k = 0; k < H_DIM; ++k) {
            float hv = hbuf[cur][k];
            hr += Whh[(long)j * H_DIM + k] * hv;
            hz += Whh[(long)(H_DIM + j) * H_DIM + k] * hv;
            hn += Whh[(long)(2 * H_DIM + j) * H_DIM + k] * hv;
        }
        float r = sigmoidf_(ir + hr);
        float z = sigmoidf_(iz + hz);
        float n = tanhf(in_ + r * hn);
        hbuf[nxt][j] = (1.f - z) * n + z * hbuf[cur][j];
        __syncthreads();
    }
    hc_out[j] = hbuf[0][j];
}

// ---------------------------------------------------------------------------
// Fused GRU gate update: consumes gi/gh pre-activations, writes h (f32 + bf16)
// ---------------------------------------------------------------------------
__global__ void gru_update(const float* __restrict__ gi, const float* __restrict__ gh,
                           float* __restrict__ h, bf16_t* __restrict__ hb) {
    long idx = (long)blockIdx.x * blockDim.x + threadIdx.x;
    if (idx >= (long)NS * H_DIM) return;
    int i = (int)(idx >> 8), j = (int)(idx & 255);
    const float* gir = gi + (long)i * G3;
    const float* ghr = gh + (long)i * G3;
    float r  = sigmoidf_(gir[j] + ghr[j]);
    float z  = sigmoidf_(gir[H_DIM + j] + ghr[H_DIM + j]);
    float n  = tanhf(gir[2 * H_DIM + j] + r * ghr[2 * H_DIM + j]);
    float hp = h[idx];
    float hv = (1.f - z) * n + z * hp;
    h[idx]  = hv;
    hb[idx] = (bf16_t)hv;
}

// ---------------------------------------------------------------------------
// Gated fusion: g = sigmoid(HS@Wg_s^T + hc@Wg_c^T); HSg = g*HS + (1-g)*hc
// ---------------------------------------------------------------------------
__global__ void gated_fusion(const float* __restrict__ HS, const float* __restrict__ hc,
                             const float* __restrict__ Wg_s, const float* __restrict__ Wg_c,
                             bf16_t* __restrict__ HSg_b, bf16_t* __restrict__ HSgT_b,
                             bf16_t* __restrict__ cat1) {
    int i = blockIdx.x * blockDim.x + threadIdx.x;
    if (i >= NS) return;
    const float* hs = HS + (long)i * H_DIM;
    float acc = 0.f;
    for (int k = 0; k < H_DIM; ++k) acc += hs[k] * Wg_s[k] + hc[k] * Wg_c[k];
    float g = sigmoidf_(acc);
    for (int k = 0; k < H_DIM; ++k) {
        float v = g * hs[k] + (1.f - g) * hc[k];
        HSg_b[(long)i * H_DIM + k] = (bf16_t)v;
        HSgT_b[(long)k * NS + i]   = (bf16_t)v;
        cat1[(long)i * 512 + k]    = (bf16_t)hs[k];
    }
}

// ---------------------------------------------------------------------------
// Row softmax over the 2048x2048 score matrix -> bf16 attention weights.
// (Per-row additive term HSg@Was^T+bas dropped: softmax is shift-invariant.)
// ---------------------------------------------------------------------------
__global__ __launch_bounds__(256)
void softmax_rows(float* __restrict__ S, bf16_t* __restrict__ Cb) {
    __shared__ float red[256];
    int row = blockIdx.x, tid = threadIdx.x;
    float* srow = S + (long)row * NS;
    float m = -3.4e38f;
    for (int j = tid; j < NS; j += 256) m = fmaxf(m, srow[j]);
    red[tid] = m; __syncthreads();
    for (int s = 128; s > 0; s >>= 1) { if (tid < s) red[tid] = fmaxf(red[tid], red[tid + s]); __syncthreads(); }
    m = red[0]; __syncthreads();
    float sum = 0.f;
    for (int j = tid; j < NS; j += 256) { float e = expf(srow[j] - m); srow[j] = e; sum += e; }
    red[tid] = sum; __syncthreads();
    for (int s = 128; s > 0; s >>= 1) { if (tid < s) red[tid] += red[tid + s]; __syncthreads(); }
    float inv = 1.f / red[0];
    bf16_t* crow = Cb + (long)row * NS;
    for (int j = tid; j < NS; j += 256) crow[j] = (bf16_t)(srow[j] * inv);
}

// ---------------------------------------------------------------------------
// H_tilde = tanh(pre1); build joint features cat2 = [hc, Ht, hc*Ht, |hc-Ht|]
// ---------------------------------------------------------------------------
__global__ void tanh_cat2(const float* __restrict__ pre, const float* __restrict__ hc,
                          bf16_t* __restrict__ cat2) {
    long idx = (long)blockIdx.x * blockDim.x + threadIdx.x;
    if (idx >= (long)NS * H_DIM) return;
    int i = (int)(idx >> 8), j = (int)(idx & 255);
    float t = tanhf(pre[idx]);
    float c = hc[j];
    bf16_t* row = cat2 + (long)i * 1024;
    row[j]             = (bf16_t)c;
    row[H_DIM + j]     = (bf16_t)t;
    row[2 * H_DIM + j] = (bf16_t)(c * t);
    row[3 * H_DIM + j] = (bf16_t)fabsf(c - t);
}

__global__ void tanh_inplace(float* __restrict__ x, long n) {
    long idx = (long)blockIdx.x * blockDim.x + threadIdx.x;
    if (idx < n) x[idx] = tanhf(x[idx]);
}

// ---------------------------------------------------------------------------
// Entailment attention + final classifier (single workgroup).
// ---------------------------------------------------------------------------
__global__ __launch_bounds__(256)
void entail_final(const float* __restrict__ Hcs, const float* __restrict__ Wae,
                  const float* __restrict__ bae, const float* __restrict__ Wf,
                  const float* __restrict__ bfin, float* __restrict__ out) {
    __shared__ float ae[NS];
    __shared__ float red[256];
    __shared__ float hsum[H_DIM];
    int tid = threadIdx.x;
    float lmax = -3.4e38f;
    for (int i = tid; i < NS; i += 256) {
        const float* hrow = Hcs + (long)i * H_DIM;
        float a = bae[0];
        for (int k = 0; k < H_DIM; ++k) a += hrow[k] * Wae[k];
        a = tanhf(a);
        ae[i] = a;
        lmax = fmaxf(lmax, a);
    }
    red[tid] = lmax; __syncthreads();
    for (int s = 128; s > 0; s >>= 1) { if (tid < s) red[tid] = fmaxf(red[tid], red[tid + s]); __syncthreads(); }
    float m = red[0]; __syncthreads();
    float lsum = 0.f;
    for (int i = tid; i < NS; i += 256) { float e = expf(ae[i] - m); ae[i] = e; lsum += e; }
    red[tid] = lsum; __syncthreads();
    for (int s = 128; s > 0; s >>= 1) { if (tid < s) red[tid] += red[tid + s]; __syncthreads(); }
    float inv = 1.f / red[0]; __syncthreads();
    float acc = 0.f;
    for (int i = 0; i < NS; ++i) acc += ae[i] * Hcs[(long)i * H_DIM + tid];
    hsum[tid] = acc * inv;
    __syncthreads();
    if (tid < 2) {
        float o = bfin[tid];
        for (int k = 0; k < H_DIM; ++k) o += hsum[k] * Wf[tid * H_DIM + k];
        out[tid] = sigmoidf_(o);
    }
}

// ---------------------------------------------------------------------------
extern "C" void kernel_launch(void* const* d_in, const int* in_sizes, int n_in,
                              void* d_out, int out_size, void* d_ws, size_t ws_size,
                              hipStream_t stream) {
    const float* claim     = (const float*)d_in[0];
    const float* sentences = (const float*)d_in[1];
    const float* Wih_c = (const float*)d_in[2];
    const float* Whh_c = (const float*)d_in[3];
    const float* bih_c = (const float*)d_in[4];
    const float* bhh_c = (const float*)d_in[5];
    const float* Wih_s = (const float*)d_in[6];
    const float* Whh_s = (const float*)d_in[7];
    const float* bih_s = (const float*)d_in[8];
    const float* bhh_s = (const float*)d_in[9];
    const float* Wg_s  = (const float*)d_in[10];
    const float* Wg_c  = (const float*)d_in[11];
    const float* Wac   = (const float*)d_in[12];
    const float* bac   = (const float*)d_in[13];
    // d_in[14]=Was, d_in[15]=bas: dropped (constant per softmax row)
    const float* Wext  = (const float*)d_in[16];
    const float* bext  = (const float*)d_in[17];
    const float* Wj    = (const float*)d_in[18];
    const float* Wae   = (const float*)d_in[19];
    const float* bae   = (const float*)d_in[20];
    const float* Wf    = (const float*)d_in[21];
    const float* bfin  = (const float*)d_in[22];

    // ---- workspace bump allocator (~60 MB total) ----
    char* ws = (char*)d_ws;
    size_t off = 0;
    auto alloc = [&](size_t bytes) -> char* {
        off = (off + 255) & ~(size_t)255;
        char* p = ws + off;
        off += bytes;
        return p;
    };
    float*  hc      = (float*) alloc(H_DIM * 4);
    float*  h       = (float*) alloc((size_t)NS * H_DIM * 4);
    bf16_t* hb      = (bf16_t*)alloc((size_t)NS * H_DIM * 2);
    float*  gi      = (float*) alloc((size_t)NS * G3 * 4);
    float*  gh      = (float*) alloc((size_t)NS * G3 * 4);
    bf16_t* Wih_b   = (bf16_t*)alloc((size_t)G3 * E_PAD * 2);
    bf16_t* Whh_b   = (bf16_t*)alloc((size_t)G3 * H_DIM * 2);
    bf16_t* Wac_b   = (bf16_t*)alloc((size_t)H_DIM * H_DIM * 2);
    bf16_t* Wext_b  = (bf16_t*)alloc((size_t)H_DIM * 512 * 2);
    bf16_t* Wj_b    = (bf16_t*)alloc((size_t)H_DIM * 1024 * 2);
    bf16_t* HSg_b   = (bf16_t*)alloc((size_t)NS * H_DIM * 2);
    bf16_t* HSgT_b  = (bf16_t*)alloc((size_t)H_DIM * NS * 2);
    float*  P       = (float*) alloc((size_t)NS * H_DIM * 4);
    bf16_t* P_b     = (bf16_t*)alloc((size_t)NS * H_DIM * 2);
    float*  scores  = (float*) alloc((size_t)NS * NS * 4);
    bf16_t* coh_b   = (bf16_t*)alloc((size_t)NS * NS * 2);
    float*  Hapo    = (float*) alloc((size_t)NS * H_DIM * 4);
    bf16_t* cat1    = (bf16_t*)alloc((size_t)NS * 512 * 2);
    float*  pre1    = (float*) alloc((size_t)NS * H_DIM * 4);
    bf16_t* cat2    = (bf16_t*)alloc((size_t)NS * 1024 * 2);
    float*  pre2    = (float*) alloc((size_t)NS * H_DIM * 4); // becomes H_c_s

    auto nblk = [](long n) { return (int)((n + 255) / 256); };

    // h0 = 0 (f32 and bf16 mirrors)
    hipMemsetAsync(h,  0, (size_t)NS * H_DIM * 4, stream);
    hipMemsetAsync(hb, 0, (size_t)NS * H_DIM * 2, stream);

    // weight casts (Wih_s K zero-padded 300 -> 320)
    cast_f32_bf16<<<nblk((long)G3 * E_PAD), 256, 0, stream>>>(Wih_s, E_DIM, Wih_b, E_PAD, 0, G3, E_DIM, E_PAD);
    cast_f32_bf16<<<nblk((long)G3 * H_DIM), 256, 0, stream>>>(Whh_s, H_DIM, Whh_b, H_DIM, 0, G3, H_DIM, H_DIM);
    cast_f32_bf16<<<nblk((long)H_DIM * H_DIM), 256, 0, stream>>>(Wac,  H_DIM, Wac_b,  H_DIM, 0, H_DIM, H_DIM, H_DIM);
    cast_f32_bf16<<<nblk((long)H_DIM * 512),  256, 0, stream>>>(Wext, 512,  Wext_b, 512,  0, H_DIM, 512, 512);
    cast_f32_bf16<<<nblk((long)H_DIM * 1024), 256, 0, stream>>>(Wj,   1024, Wj_b,   1024, 0, H_DIM, 1024, 1024);

    // claim encoder (1-row GRU, scalar)
    claim_gru<<<1, 256, 0, stream>>>(claim, Wih_c, Whh_c, bih_c, bhh_c, hc);

    // sentence GRU: 64 sequential steps. gi GEMM converts fp32 x_t in-register
    // (A = sentences + t*300, row stride 64*300, tail-tile guard k<300);
    // gh GEMM uses the bf16 hidden mirror written by the previous gate update.
    const dim3 gruGrid(G3 / 64, NS / 128);
    for (int t = 0; t < TLEN; ++t) {
        gemm_wmma<true><<<gruGrid, 128, 0, stream>>>(
            sentences + (long)t * E_DIM, (long)TLEN * E_DIM, E_DIM,
            Wih_b, E_PAD, bih_s, gi, G3, E_PAD);
        gemm_wmma<false><<<gruGrid, 128, 0, stream>>>(
            hb, H_DIM, H_DIM, Whh_b, H_DIM, bhh_s, gh, G3, H_DIM);
        gru_update<<<nblk((long)NS * H_DIM), 256, 0, stream>>>(gi, gh, h, hb);
    }

    // gated fusion (also fills cat1[:,0:256] = bf16(HS) and HSg^T)
    gated_fusion<<<nblk(NS), 256, 0, stream>>>(h, hc, Wg_s, Wg_c, HSg_b, HSgT_b, cat1);

    // P = HSg @ Wac^T + bac
    gemm_wmma<false><<<dim3(H_DIM / 64, NS / 128), 128, 0, stream>>>(
        HSg_b, H_DIM, H_DIM, Wac_b, H_DIM, bac, P, H_DIM, H_DIM);
    cast_f32_bf16<<<nblk((long)NS * H_DIM), 256, 0, stream>>>(P, H_DIM, P_b, H_DIM, 0, NS, H_DIM, H_DIM);

    // scores = P @ HSg^T   (2048 x 2048; row-constant bias term dropped)
    gemm_wmma<false><<<dim3(NS / 64, NS / 128), 128, 0, stream>>>(
        P_b, H_DIM, H_DIM, HSg_b, H_DIM, nullptr, scores, NS, H_DIM);

    // coh = softmax_rows(scores) -> bf16
    softmax_rows<<<NS, 256, 0, stream>>>(scores, coh_b);

    // H_apo = coh @ HSg  (via HSg^T as the "weight")
    gemm_wmma<false><<<dim3(H_DIM / 64, NS / 128), 128, 0, stream>>>(
        coh_b, NS, NS, HSgT_b, NS, nullptr, Hapo, H_DIM, NS);
    // cat1[:,256:512] = bf16(H_apo)
    cast_f32_bf16<<<nblk((long)NS * H_DIM), 256, 0, stream>>>(Hapo, H_DIM, cat1, 512, 256, NS, H_DIM, H_DIM);

    // pre1 = cat1 @ Wext^T + bext ; H_tilde = tanh(pre1) fused into cat2 build
    gemm_wmma<false><<<dim3(H_DIM / 64, NS / 128), 128, 0, stream>>>(
        cat1, 512, 512, Wext_b, 512, bext, pre1, H_DIM, 512);
    tanh_cat2<<<nblk((long)NS * H_DIM), 256, 0, stream>>>(pre1, hc, cat2);

    // pre2 = cat2 @ Wj^T ; H_c_s = tanh(pre2)
    gemm_wmma<false><<<dim3(H_DIM / 64, NS / 128), 128, 0, stream>>>(
        cat2, 1024, 1024, Wj_b, 1024, nullptr, pre2, H_DIM, 1024);
    tanh_inplace<<<nblk((long)NS * H_DIM), 256, 0, stream>>>(pre2, (long)NS * H_DIM);

    // entailment attention + classifier -> 2 outputs
    entail_final<<<1, 256, 0, stream>>>(pre2, Wae, bae, Wf, bfin, (float*)d_out);
}
